// AxialBlock_4157528343341
// MI455X (gfx1250) — matrix operand order repaired and checked by reading, not compiled, not run on hardware
//
#include <hip/hip_runtime.h>

// ---------------------------------------------------------------------------
// CDNA5 / gfx1250 axial-attention block.
// All matmuls (QKV projections, Q.K^T, P.V, output projection) run on
// v_wmma_f32_16x16x32_bf16 (wave32). fp32 inputs are cast once to bf16;
// weights are cast + transposed once so all GEMM staging is contiguous.
// GEMM K-loop is double-buffered with GLOBAL_LOAD_ASYNC_TO_LDS_B128
// (s_wait_asynccnt sync); falls back to plain b128 copies if absent.
// ---------------------------------------------------------------------------

typedef unsigned short u16s;
typedef __bf16 bf16;
typedef bf16  v16bf __attribute__((ext_vector_type(16)));
typedef float v8f   __attribute__((ext_vector_type(8)));

union Frag32B { v16bf v; uint4 q[2]; };

__device__ __forceinline__ u16s f2bf(float f) {
  unsigned int x = __float_as_uint(f);
  x += 0x7FFFu + ((x >> 16) & 1u);          // round-to-nearest-even
  return (u16s)(x >> 16);
}

// ---- gfx1250 async global->LDS copy (ASYNCcnt) ----------------------------
#if defined(__has_builtin)
#if __has_builtin(__builtin_amdgcn_global_load_async_to_lds_b128) && \
    __has_builtin(__builtin_amdgcn_s_wait_asynccnt)
#define ATHENA_ASYNC 1
#endif
#endif

#ifdef ATHENA_ASYNC
typedef int v4i_vs __attribute__((vector_size(16)));            // GCC-style V4i
typedef __attribute__((address_space(1))) v4i_vs* as1_v4ip;     // global ptr
typedef __attribute__((address_space(3))) v4i_vs* as3_v4ip;     // LDS ptr
#endif

__device__ __forceinline__ void cp16_g2l(u16s* lds, const u16s* g) {
#ifdef ATHENA_ASYNC
  // flat LDS addr [31:0] is the AS3 byte offset (ISA aperture rule);
  // global generic pointers share the AS1 value.
  __builtin_amdgcn_global_load_async_to_lds_b128(
      (as1_v4ip)(uintptr_t)g,
      (as3_v4ip)(unsigned int)(uintptr_t)lds,
      0, 0);
#else
  *(uint4*)lds = *(const uint4*)g;
#endif
}

#ifdef ATHENA_ASYNC
#define ASYNC_WAIT(n) __builtin_amdgcn_s_wait_asynccnt(n)
#else
#define ASYNC_WAIT(n) ((void)0)
#endif

// ---------------------------------------------------------------------------
// fp32 -> bf16 cast, 4 elements/thread (activations)
// ---------------------------------------------------------------------------
__global__ __launch_bounds__(256)
void cast_f32_bf16(const float* __restrict__ s, u16s* __restrict__ d, long n4) {
  long i = (long)blockIdx.x * blockDim.x + threadIdx.x;
  if (i >= n4) return;
  float4 f = ((const float4*)s)[i];
  uint2 o;
  o.x = (unsigned)f2bf(f.x) | ((unsigned)f2bf(f.y) << 16);
  o.y = (unsigned)f2bf(f.z) | ((unsigned)f2bf(f.w) << 16);
  ((uint2*)d)[i] = o;
}

// ---------------------------------------------------------------------------
// fp32 [512][512] -> bf16 transposed [n][k] (weights, once per matrix).
// 64x64 tile through LDS; coalesced read and write.
// ---------------------------------------------------------------------------
__global__ __launch_bounds__(256)
void castT_f32_bf16(const float* __restrict__ W, u16s* __restrict__ Wt) {
  __shared__ float T[64][65];
  const int tx = threadIdx.x & 63, ty = threadIdx.x >> 6;
  const int k0 = blockIdx.x << 6, n0 = blockIdx.y << 6;
#pragma unroll
  for (int r = ty; r < 64; r += 4)
    T[r][tx] = W[(size_t)(k0 + r) * 512 + n0 + tx];
  __syncthreads();
#pragma unroll
  for (int r = ty; r < 64; r += 4)
    Wt[(size_t)(n0 + r) * 512 + k0 + tx] = f2bf(T[tx][r]);
}

// ---------------------------------------------------------------------------
// Shared GEMM tile staging: A row-major [m][k], Wt row-major [n][k].
// 6 x 16B async copies per thread per 128x64 / 64x64 tile pair.
// ---------------------------------------------------------------------------
__device__ __forceinline__ void stage_tile(const u16s* __restrict__ A,
                                           const u16s* __restrict__ Wt,
                                           size_t bm, int bn, int kt, int tid,
                                           u16s (*As)[72], u16s (*Bs)[72]) {
#pragma unroll
  for (int c = 0; c < 4; ++c) {
    int e = tid + (c << 8);
    int m = e >> 3, kk = (e & 7) << 3;
    cp16_g2l(&As[m][kk], &A[(bm + m) * 512 + kt + kk]);
  }
#pragma unroll
  for (int c = 0; c < 2; ++c) {
    int e = tid + (c << 8);
    int n = e >> 3, kk = (e & 7) << 3;
    cp16_g2l(&Bs[n][kk], &Wt[(size_t)(bn + n) * 512 + kt + kk]);
  }
}

// ---------------------------------------------------------------------------
// GEMM: C[M,512] = A[M,512](bf16) x W[512,512] (Wt = W^T, bf16), bf16 out.
// Block 256 thr = 8 waves (4x2); wave computes 32x32 via 2x2 WMMA tiles.
// BM=128, BN=64, BK=64, double-buffered LDS (rows padded to 72 halfwords).
// Epilogue goes through LDS so global stores are b128.
// ---------------------------------------------------------------------------
__global__ __launch_bounds__(256)
void gemm512_bf16(const u16s* __restrict__ A, const u16s* __restrict__ Wt,
                  u16s* __restrict__ Cb) {
  __shared__ __align__(16) u16s As[2][128][72];
  __shared__ __align__(16) u16s Bs[2][64][72];
  const int tid  = threadIdx.x;
  const size_t bm = (size_t)blockIdx.x * 128;
  const int bn   = blockIdx.y * 64;
  const int wave = tid >> 5, lane = tid & 31;
  const int wm   = (wave >> 1) << 5, wn = (wave & 1) << 5;
  const int lg   = lane & 15;
  const bool hi  = lane >= 16;

  v8f acc[2][2] = {};

  stage_tile(A, Wt, bm, bn, 0, tid, As[0], Bs[0]);
  for (int kt = 0; kt < 512; kt += 64) {
    const int cur = (kt >> 6) & 1;
    if (kt + 64 < 512) {
      stage_tile(A, Wt, bm, bn, kt + 64, tid, As[cur ^ 1], Bs[cur ^ 1]);
      ASYNC_WAIT(6);           // previous tile's copies have landed
    } else {
      ASYNC_WAIT(0);
    }
    __syncthreads();
#pragma unroll
    for (int ks = 0; ks < 2; ++ks) {
      Frag32B a[2], b[2];
      const int akb = (ks << 5) + (hi ? 8 : 0);   // A: lo K0-7/16-23, hi +8
      const int bkb = (ks << 5) + (hi ? 8 : 0);   // B: lo K0-7/16-23, hi +8
#pragma unroll
      for (int tm = 0; tm < 2; ++tm) {
        const int m = wm + (tm << 4) + lg;
        a[tm].q[0] = *(const uint4*)&As[cur][m][akb];
        a[tm].q[1] = *(const uint4*)&As[cur][m][akb + 16];
      }
#pragma unroll
      for (int tn = 0; tn < 2; ++tn) {
        const int n = wn + (tn << 4) + lg;
        b[tn].q[0] = *(const uint4*)&Bs[cur][n][bkb];
        b[tn].q[1] = *(const uint4*)&Bs[cur][n][bkb + 16];
      }
#pragma unroll
      for (int tm = 0; tm < 2; ++tm)
#pragma unroll
        for (int tn = 0; tn < 2; ++tn)
          acc[tm][tn] = __builtin_amdgcn_wmma_f32_16x16x32_bf16(
              false, a[tm].v, false, b[tn].v, (short)0, acc[tm][tn], false, false);
    }
    __syncthreads();
  }

  // epilogue: C-layout -> LDS -> coalesced b128 stores
  u16s (*Cs)[72] = As[0];
#pragma unroll
  for (int tm = 0; tm < 2; ++tm)
#pragma unroll
    for (int tn = 0; tn < 2; ++tn)
#pragma unroll
      for (int r = 0; r < 8; ++r)
        Cs[wm + (tm << 4) + (hi ? 8 + r : r)][wn + (tn << 4) + lg] =
            f2bf(acc[tm][tn][r]);
  __syncthreads();
#pragma unroll
  for (int c = 0; c < 4; ++c) {
    int e = tid + (c << 8);
    int m = e >> 3, cc = (e & 7) << 3;
    *(uint4*)&Cb[(bm + m) * 512 + bn + cc] = *(const uint4*)&Cs[m][cc];
  }
}

// ---------------------------------------------------------------------------
// Projection GEMM: Out[M,512] (+)= A x Wo + bias ; f32 out, float4 stores.
// ---------------------------------------------------------------------------
__global__ __launch_bounds__(256)
void gemm512_proj(const u16s* __restrict__ A, const u16s* __restrict__ Wt,
                  const float* __restrict__ bias, float* __restrict__ Out,
                  int addTo) {
  __shared__ __align__(16) u16s As[2][128][72];
  __shared__ __align__(16) u16s Bs[2][64][72];
  const int tid  = threadIdx.x;
  const size_t bm = (size_t)blockIdx.x * 128;
  const int bn   = blockIdx.y * 64;
  const int wave = tid >> 5, lane = tid & 31;
  const int wm   = (wave >> 1) << 5, wn = (wave & 1) << 5;
  const int lg   = lane & 15;
  const bool hi  = lane >= 16;

  v8f acc[2][2] = {};

  stage_tile(A, Wt, bm, bn, 0, tid, As[0], Bs[0]);
  for (int kt = 0; kt < 512; kt += 64) {
    const int cur = (kt >> 6) & 1;
    if (kt + 64 < 512) {
      stage_tile(A, Wt, bm, bn, kt + 64, tid, As[cur ^ 1], Bs[cur ^ 1]);
      ASYNC_WAIT(6);
    } else {
      ASYNC_WAIT(0);
    }
    __syncthreads();
#pragma unroll
    for (int ks = 0; ks < 2; ++ks) {
      Frag32B a[2], b[2];
      const int akb = (ks << 5) + (hi ? 8 : 0);
      const int bkb = (ks << 5) + (hi ? 8 : 0);
#pragma unroll
      for (int tm = 0; tm < 2; ++tm) {
        const int m = wm + (tm << 4) + lg;
        a[tm].q[0] = *(const uint4*)&As[cur][m][akb];
        a[tm].q[1] = *(const uint4*)&As[cur][m][akb + 16];
      }
#pragma unroll
      for (int tn = 0; tn < 2; ++tn) {
        const int n = wn + (tn << 4) + lg;
        b[tn].q[0] = *(const uint4*)&Bs[cur][n][bkb];
        b[tn].q[1] = *(const uint4*)&Bs[cur][n][bkb + 16];
      }
#pragma unroll
      for (int tm = 0; tm < 2; ++tm)
#pragma unroll
        for (int tn = 0; tn < 2; ++tn)
          acc[tm][tn] = __builtin_amdgcn_wmma_f32_16x16x32_bf16(
              false, a[tm].v, false, b[tn].v, (short)0, acc[tm][tn], false, false);
    }
    __syncthreads();
  }

  // epilogue: +bias, (+=) through LDS, float4 global stores
  float (*Cf)[68] = (float (*)[68]) & As[0][0][0];  // 128*68*4 <= sizeof(As)
#pragma unroll
  for (int tm = 0; tm < 2; ++tm)
#pragma unroll
    for (int tn = 0; tn < 2; ++tn)
#pragma unroll
      for (int r = 0; r < 8; ++r) {
        int lr = wm + (tm << 4) + (hi ? 8 + r : r);
        int lc = wn + (tn << 4) + lg;
        Cf[lr][lc] = acc[tm][tn][r] + bias[bn + lc];
      }
  __syncthreads();
#pragma unroll
  for (int c = 0; c < 8; ++c) {
    int e = tid + (c << 8);
    int m = e >> 4, cc = (e & 15) << 2;
    size_t o = (bm + m) * 512 + bn + cc;
    float4 v = *(const float4*)&Cf[m][cc];
    if (addTo) {
      float4 g = *(const float4*)&Out[o];
      v.x += g.x; v.y += g.y; v.z += g.z; v.w += g.w;
    }
    *(float4*)&Out[o] = v;
  }
}

// ---------------------------------------------------------------------------
// WMMA axial attention for one (sequence, head).
// L = 64: block = 4 waves (one 16-query chunk each). L = 16: block = 1 wave,
// K padded to 32 with zeros so 16x16x32 WMMA applies unchanged.
// ---------------------------------------------------------------------------
template <int L>
__global__ __launch_bounds__(32 * (L / 16))
void attn_kernel(const u16s* __restrict__ Qg, const u16s* __restrict__ Kg,
                 const u16s* __restrict__ Vg, u16s* __restrict__ Og,
                 int seqDiv, int sOuter, int sInner, int eStride) {
  constexpr int NT   = L / 16;             // 16x16 tile count along L
  constexpr int PK   = (L == 16) ? 32 : L; // padded K for the P.V GEMM
  constexpr int VSTR = PK + 8;
  constexpr int PSTR = PK + 8;
  constexpr int NTHR = 32 * NT;

  __shared__ __align__(16) u16s Qs[L][72];
  __shared__ __align__(16) u16s Ks[L][72];
  __shared__ __align__(16) u16s Vt[64][VSTR];
  __shared__ __align__(16) u16s Ps[NT][16][PSTR];

  const int tid  = threadIdx.x;
  const int s    = blockIdx.x;
  const int head = blockIdx.y;
  const size_t base = (size_t)(s / seqDiv) * sOuter + (size_t)(s % seqDiv) * sInner;
  const int hoff = head * 64;
  const int wq   = tid >> 5, lane = tid & 31;
  const int lg   = lane & 15;
  const bool hi  = lane >= 16;

  // ---- stage Q, K (async, row-major) and V (transposed scatter) ----------
  for (int e = tid; e < L * 8; e += NTHR) {
    int j = e >> 3, cc = (e & 7) << 3;
    size_t g = (base + (size_t)j * eStride) * 512 + hoff + cc;
    cp16_g2l(&Qs[j][cc], &Qg[g]);
    cp16_g2l(&Ks[j][cc], &Kg[g]);
    uint4 dv = *(const uint4*)&Vg[g];
    const u16s* ds = (const u16s*)&dv;
#pragma unroll
    for (int i = 0; i < 8; ++i) Vt[cc + i][j] = ds[i];
  }
  if (L == 16) {  // zero-pad K dimension 16..31 of V^T
    for (int e = tid; e < 64 * 16; e += NTHR) Vt[e & 63][16 + (e >> 6)] = (u16s)0;
  }
  ASYNC_WAIT(0);
  __syncthreads();

  // ---- S = Q.K^T, wave handles 16 queries x L keys -----------------------
  v8f sc[NT] = {};
#pragma unroll
  for (int ks = 0; ks < 2; ++ks) {           // dk = 64 -> 2 k-steps of 32
    Frag32B a;
    const int akb = (ks << 5) + (hi ? 8 : 0);
    const int m   = (wq << 4) + lg;
    a.q[0] = *(const uint4*)&Qs[m][akb];
    a.q[1] = *(const uint4*)&Qs[m][akb + 16];
#pragma unroll
    for (int nt = 0; nt < NT; ++nt) {
      Frag32B b;
      const int n   = (nt << 4) + lg;
      const int bkb = (ks << 5) + (hi ? 8 : 0);
      b.q[0] = *(const uint4*)&Ks[n][bkb];
      b.q[1] = *(const uint4*)&Ks[n][bkb + 16];
      sc[nt] = __builtin_amdgcn_wmma_f32_16x16x32_bf16(
          false, a.v, false, b.v, (short)0, sc[nt], false, false);
    }
  }

  // ---- softmax over each score row (16 lanes x NT tiles) -----------------
  const float scale = 0.125f;  // 1/sqrt(64)
  float mx[8], sm[8];
#pragma unroll
  for (int r = 0; r < 8; ++r) {
    float m = -3.0e38f;
#pragma unroll
    for (int nt = 0; nt < NT; ++nt) {
      float v = sc[nt][r] * scale;
      sc[nt][r] = v;
      m = fmaxf(m, v);
    }
    mx[r] = m;
  }
#pragma unroll
  for (int off = 8; off > 0; off >>= 1)
#pragma unroll
    for (int r = 0; r < 8; ++r)
      mx[r] = fmaxf(mx[r], __shfl_xor(mx[r], off, 32));
#pragma unroll
  for (int r = 0; r < 8; ++r) {
    float t = 0.f;
#pragma unroll
    for (int nt = 0; nt < NT; ++nt) {
      float e = __expf(sc[nt][r] - mx[r]);
      sc[nt][r] = e;
      t += e;
    }
    sm[r] = t;
  }
#pragma unroll
  for (int off = 8; off > 0; off >>= 1)
#pragma unroll
    for (int r = 0; r < 8; ++r)
      sm[r] += __shfl_xor(sm[r], off, 32);

  // ---- write normalized P (bf16) to LDS in A-matrix layout ---------------
  if (L == 16) {  // zero-pad K 16..31 of P
    for (int e = lane; e < 16 * 16; e += 32) Ps[wq][e >> 4][16 + (e & 15)] = (u16s)0;
  }
#pragma unroll
  for (int nt = 0; nt < NT; ++nt)
#pragma unroll
    for (int r = 0; r < 8; ++r) {
      int mrow = hi ? 8 + r : r;
      Ps[wq][mrow][(nt << 4) + lg] = f2bf(sc[nt][r] / sm[r]);
    }
  // same-wave DS ops are in-order: no barrier needed before re-reading Ps.

  // ---- O = P.V : M=16, N=64, K=PK ----------------------------------------
  v8f ob[4] = {};
#pragma unroll
  for (int ks = 0; ks < PK / 32; ++ks) {
    Frag32B a;
    const int akb = (ks << 5) + (hi ? 8 : 0);
    a.q[0] = *(const uint4*)&Ps[wq][lg][akb];
    a.q[1] = *(const uint4*)&Ps[wq][lg][akb + 16];
#pragma unroll
    for (int nt = 0; nt < 4; ++nt) {
      Frag32B b;
      const int n   = (nt << 4) + lg;
      const int bkb = (ks << 5) + (hi ? 8 : 0);
      b.q[0] = *(const uint4*)&Vt[n][bkb];
      b.q[1] = *(const uint4*)&Vt[n][bkb + 16];
      ob[nt] = __builtin_amdgcn_wmma_f32_16x16x32_bf16(
          false, a.v, false, b.v, (short)0, ob[nt], false, false);
    }
  }

  // ---- store O (bf16) back to the Q buffer slice -------------------------
#pragma unroll
  for (int nt = 0; nt < 4; ++nt)
#pragma unroll
    for (int r = 0; r < 8; ++r) {
      int mrow = hi ? 8 + r : r;
      size_t grow = base + (size_t)((wq << 4) + mrow) * eStride;
      Og[grow * 512 + hoff + (nt << 4) + lg] = f2bf(ob[nt][r]);
    }
}

// ---------------------------------------------------------------------------
// Host-side launch.  Inputs: [0]=x, then per branch (w,h,t):
// Wq, Wk, Wv, Wo, bo.  Output: f32 [2,16,64,64,512].
// ---------------------------------------------------------------------------
extern "C" void kernel_launch(void* const* d_in, const int* in_sizes, int n_in,
                              void* d_out, int out_size, void* d_ws, size_t ws_size,
                              hipStream_t stream) {
  (void)in_sizes; (void)n_in; (void)out_size; (void)ws_size;
  constexpr size_t NTOK = 2ull * 16 * 64 * 64;  // 131072 tokens
  constexpr size_t Cc   = 512;

  char* ws = (char*)d_ws;
  size_t off = 0;
  auto carve = [&](size_t bytes) -> void* {
    void* p = ws + off;
    off += (bytes + 255) & ~(size_t)255;
    return p;
  };
  u16s* wbf[12];  // transposed bf16 weights [n][k]
  for (int i = 0; i < 12; ++i) wbf[i] = (u16s*)carve(Cc * Cc * 2);
  u16s* xb = (u16s*)carve(NTOK * Cc * 2);
  u16s* qb = (u16s*)carve(NTOK * Cc * 2);   // also holds attention output O
  u16s* kb = (u16s*)carve(NTOK * Cc * 2);
  u16s* vb = (u16s*)carve(NTOK * Cc * 2);

  // casts: x -> bf16, 12 weight matrices -> transposed bf16
  {
    long n4 = (long)(NTOK * Cc / 4);
    cast_f32_bf16<<<dim3((unsigned)((n4 + 255) / 256)), 256, 0, stream>>>(
        (const float*)d_in[0], xb, n4);
  }
  for (int br = 0; br < 3; ++br)
    for (int p = 0; p < 4; ++p)
      castT_f32_bf16<<<dim3(8, 8), 256, 0, stream>>>(
          (const float*)d_in[1 + br * 5 + p], wbf[br * 4 + p]);

  // axial geometry: base = (s/seqDiv)*sOuter + (s%seqDiv)*sInner, stride eStr
  const int seqDiv[3] = {1,    64,   4096};
  const int sOuter[3] = {64,   4096, 65536};
  const int sInner[3] = {0,    1,    1};
  const int eStr[3]   = {1,    64,   4096};
  const int nSeq[3]   = {2048, 2048, 8192};

  const dim3 ggrid((unsigned)(NTOK / 128), 8);

  for (int br = 0; br < 3; ++br) {
    gemm512_bf16<<<ggrid, 256, 0, stream>>>(xb, wbf[br * 4 + 0], qb);
    gemm512_bf16<<<ggrid, 256, 0, stream>>>(xb, wbf[br * 4 + 1], kb);
    gemm512_bf16<<<ggrid, 256, 0, stream>>>(xb, wbf[br * 4 + 2], vb);

    if (br < 2)  // w and h axes: L = 64
      attn_kernel<64><<<dim3(nSeq[br], 8), 128, 0, stream>>>(
          qb, kb, vb, qb, seqDiv[br], sOuter[br], sInner[br], eStr[br]);
    else         // t axis: L = 16
      attn_kernel<16><<<dim3(nSeq[br], 8), 32, 0, stream>>>(
          qb, kb, vb, qb, seqDiv[br], sOuter[br], sInner[br], eStr[br]);

    gemm512_proj<<<ggrid, 256, 0, stream>>>(
        qb, wbf[br * 4 + 3], (const float*)d_in[1 + br * 5 + 4],
        (float*)d_out, br != 0);
  }
}
